// SelfAttention_4647154614572
// MI455X (gfx1250) — matrix operand order repaired and checked by reading, not compiled, not run on hardware
//
#include <hip/hip_runtime.h>

// Problem constants (match reference)
#define BB 8
#define TT 1024
#define EE 256
#define HH 8
#define NE (HH * EE)   // 2048
#define BT (BB * TT)   // 8192

typedef __attribute__((ext_vector_type(16))) __bf16 v16bf;
typedef __attribute__((ext_vector_type(8)))  __bf16 v8bf;
typedef __attribute__((ext_vector_type(8)))  float  v8f;

// ---------------------------------------------------------------------------
// Fragment helpers (wave32; A/B share a contiguous k order, not the ISA table's)
//   A (16x32 bf16): lane m = lane&15; k = ((lane>=16)?16:0) + i   (contiguous, matches B)
//   B (32x16 bf16): lane n = lane&15; k = ((lane>=16)?16:0) + i   (contiguous!)
//   C (16x16 f32) : reg j -> row j + 8*(lane>=16), col = lane&15
// ---------------------------------------------------------------------------
__device__ __forceinline__ v16bf load_afrag(const __bf16* rowptr /* already + (hi?16:0) */,
                                            int k0) {
  v8bf a0 = *reinterpret_cast<const v8bf*>(rowptr + k0);
  v8bf a1 = *reinterpret_cast<const v8bf*>(rowptr + k0 + 8);
  v16bf a;
  #pragma unroll
  for (int i = 0; i < 8; ++i) { a[i] = a0[i]; a[i + 8] = a1[i]; }
  return a;
}

// ---------------------------------------------------------------------------
// Kernel 0: precision / layout prep
//   task 0: x  (f32, [BT,E])      -> xb  bf16 row-major
//   task 1-3: Wq/Wk/Wv ([E,NE])   -> Wt[z] bf16 transposed [NE,E]
//   task 4: Wo ([NE,E])           -> Wot bf16 transposed [E,NE]
// ---------------------------------------------------------------------------
__global__ __launch_bounds__(256)
void prep_kernel(const float* __restrict__ x,
                 const float* __restrict__ Wq, const float* __restrict__ Wk,
                 const float* __restrict__ Wv, const float* __restrict__ Wo,
                 __bf16* __restrict__ xb, __bf16* __restrict__ Wt,
                 __bf16* __restrict__ Wot) {
  int i = blockIdx.x * 256 + threadIdx.x;
  int task = blockIdx.y;
  if (task == 0) {
    if (i < BT * EE) xb[i] = (__bf16)x[i];
  } else if (task <= 3) {
    if (i < NE * EE) {
      const float* W = (task == 1) ? Wq : (task == 2) ? Wk : Wv;
      int n = i >> 8, k = i & 255;                 // Wt[n][k] = W[k][n]
      Wt[(size_t)(task - 1) * NE * EE + i] = (__bf16)W[(size_t)k * NE + n];
    }
  } else {
    if (i < EE * NE) {
      int n = i >> 11, kk = i & 2047;              // Wot[n][kk] = Wo[kk][n]
      Wot[i] = (__bf16)Wo[(size_t)kk * EE + n];
    }
  }
}

// ---------------------------------------------------------------------------
// Kernel 1: QKV projections.  One 16x16 output tile per wave, K = E = 256.
//   z=0 -> Qb bf16 row-major [BT,NE]
//   z=1 -> Kb bf16 row-major [BT,NE]
//   z=2 -> Vt bf16 transposed per head: Vt[((b*H+h)*E + e)*T + t]
// grid: (BT/16=512, NE/(16*8)=16, 3), block 256 (8 waves)
// ---------------------------------------------------------------------------
__global__ __launch_bounds__(256)
void qkv_kernel(const __bf16* __restrict__ xb, const __bf16* __restrict__ Wt,
                __bf16* __restrict__ Qb, __bf16* __restrict__ Kb,
                __bf16* __restrict__ Vt) {
  const int lane = threadIdx.x & 31;
  const int wave = threadIdx.x >> 5;
  const int hi = lane >> 4, ln = lane & 15;
  const int m0 = blockIdx.x * 16;
  const int nt = blockIdx.y * 8 + wave;
  const int n0 = nt * 16;
  const int z  = blockIdx.z;
  const __bf16* W = Wt + (size_t)z * NE * EE;

  const __bf16* arow = xb + (size_t)(m0 + ln) * EE + (hi ? 16 : 0);
  const __bf16* brow = W  + (size_t)(n0 + ln) * EE + (hi ? 16 : 0);

  v8f c = {};
  #pragma unroll
  for (int k0 = 0; k0 < EE; k0 += 32) {
    v16bf a = load_afrag(arow, k0);
    v16bf b = *reinterpret_cast<const v16bf*>(brow + k0);
    c = __builtin_amdgcn_wmma_f32_16x16x32_bf16(false, a, false, b,
                                                (short)0, c, false, false);
  }

  if (z < 2) {
    __bf16* dst = (z == 0) ? Qb : Kb;
    #pragma unroll
    for (int j = 0; j < 8; ++j) {
      int row = m0 + j + 8 * hi;
      dst[(size_t)row * NE + n0 + ln] = (__bf16)c[j];
    }
  } else {
    int ncol = n0 + ln, h = ncol >> 8, e = ncol & 255;
    #pragma unroll
    for (int j = 0; j < 8; ++j) {
      int row = m0 + j + 8 * hi;       // global token
      int bb = row >> 10, t = row & 1023;
      Vt[((size_t)(bb * HH + h) * EE + e) * TT + t] = (__bf16)c[j];
    }
  }
}

// ---------------------------------------------------------------------------
// Kernel 2: flash attention. One wave per (b, h, 16-query tile).
// Streams 32-key tiles: QK^T (16 WMMA) -> online softmax -> P via LDS ->
// A*V (16 WMMA) into 16x256 f32 accumulators.
// grid: (T/16=64, H=8, B=8), block 32
// ---------------------------------------------------------------------------
__global__ __launch_bounds__(32)
void attn_kernel(const __bf16* __restrict__ Qb, const __bf16* __restrict__ Kb,
                 const __bf16* __restrict__ Vt,
                 const unsigned char* __restrict__ mask,  // bool per (b,t)
                 __bf16* __restrict__ Ctx) {
  __shared__ __bf16 Plds[16 * 32];

  const int lane = threadIdx.x & 31;
  const int hi = lane >> 4, ln = lane & 15;
  const int q0 = blockIdx.x * 16;
  const int h  = blockIdx.y;
  const int b  = blockIdx.z;
  const float scale = 0.0625f;  // 1/sqrt(256)

  // Q A-fragments for all 8 e-chunks, kept resident
  v16bf qf[8];
  {
    const __bf16* qrow =
        Qb + (size_t)(b * TT + q0 + ln) * NE + h * EE + (hi ? 16 : 0);
    #pragma unroll
    for (int ec = 0; ec < 8; ++ec) qf[ec] = load_afrag(qrow, ec * 32);
  }

  v8f zero = {};
  v8f acc[16];
  #pragma unroll
  for (int i = 0; i < 16; ++i) acc[i] = zero;
  float mrun[8], lrun[8];
  #pragma unroll
  for (int j = 0; j < 8; ++j) { mrun[j] = -3.0e38f; lrun[j] = 0.f; }

  const __bf16* krowBase = Kb + (size_t)(b * TT) * NE + h * EE + (hi ? 16 : 0);
  const __bf16* vbase = Vt + (size_t)(b * HH + h) * EE * TT + (hi ? 16 : 0);
  const unsigned char* mb = mask + b * TT;

  for (int k0 = 0; k0 < TT; k0 += 32) {
    if (k0 + 32 < TT)  // gfx1250 global_prefetch_b8 for next key tile
      __builtin_prefetch((const void*)(krowBase + (size_t)(k0 + 32 + ln) * NE), 0, 1);

    // ---- scores: S[16x32] = Q (16xE) * K^T chunk ----
    v8f c0 = {}, c1 = {};
    const __bf16* kr0 = krowBase + (size_t)(k0 + ln) * NE;
    const __bf16* kr1 = krowBase + (size_t)(k0 + 16 + ln) * NE;
    #pragma unroll
    for (int ec = 0; ec < 8; ++ec) {
      v16bf b0 = *reinterpret_cast<const v16bf*>(kr0 + ec * 32);
      c0 = __builtin_amdgcn_wmma_f32_16x16x32_bf16(false, qf[ec], false, b0,
                                                   (short)0, c0, false, false);
      v16bf b1 = *reinterpret_cast<const v16bf*>(kr1 + ec * 32);
      c1 = __builtin_amdgcn_wmma_f32_16x16x32_bf16(false, qf[ec], false, b1,
                                                   (short)0, c1, false, false);
    }

    // ---- online softmax (per row j; rows live in 16-lane halves) ----
    float bias0 = mb[k0 + ln]      ? -3.0e38f : 0.f;
    float bias1 = mb[k0 + 16 + ln] ? -3.0e38f : 0.f;
    float corr[8];
    #pragma unroll
    for (int j = 0; j < 8; ++j) {
      float s0 = c0[j] * scale + bias0;
      float s1 = c1[j] * scale + bias1;
      float t = fmaxf(s0, s1);
      t = fmaxf(t, __shfl_xor(t, 1, 32));
      t = fmaxf(t, __shfl_xor(t, 2, 32));
      t = fmaxf(t, __shfl_xor(t, 4, 32));
      t = fmaxf(t, __shfl_xor(t, 8, 32));
      float mnew = fmaxf(mrun[j], t);
      float cr = __expf(mrun[j] - mnew);
      float p0 = __expf(s0 - mnew);
      float p1 = __expf(s1 - mnew);
      float rs = p0 + p1;
      rs += __shfl_xor(rs, 1, 32);
      rs += __shfl_xor(rs, 2, 32);
      rs += __shfl_xor(rs, 4, 32);
      rs += __shfl_xor(rs, 8, 32);
      lrun[j] = lrun[j] * cr + rs;
      mrun[j] = mnew;
      corr[j] = cr;
      // stash P (C-layout) into LDS row-major 16x32
      Plds[(j + 8 * hi) * 32 + ln]      = (__bf16)p0;
      Plds[(j + 8 * hi) * 32 + 16 + ln] = (__bf16)p1;
    }

    // rescale running accumulators
    #pragma unroll
    for (int nt = 0; nt < 16; ++nt)
      #pragma unroll
      for (int j = 0; j < 8; ++j) acc[nt][j] *= corr[j];

    // LDS ops are in-order within a wave; wait for stores before the b128 reads
    asm volatile("s_wait_dscnt 0x0" ::: "memory");

    // reload P in A-fragment layout
    v16bf pf;
    {
      const v8bf* pp = reinterpret_cast<const v8bf*>(&Plds[ln * 32 + hi * 16]);
      v8bf p0v = pp[0];   // k = hi*16 .. hi*16+7
      v8bf p1v = pp[1];   // k = hi*16+8 ..
      #pragma unroll
      for (int i = 0; i < 8; ++i) { pf[i] = p0v[i]; pf[i + 8] = p1v[i]; }
    }

    // ---- A*V: 16 output column tiles over e = 256 ----
    #pragma unroll
    for (int nt = 0; nt < 16; ++nt) {
      const __bf16* vr = vbase + (size_t)(nt * 16 + ln) * TT + k0;
      v16bf bv = *reinterpret_cast<const v16bf*>(vr);
      acc[nt] = __builtin_amdgcn_wmma_f32_16x16x32_bf16(false, pf, false, bv,
                                                        (short)0, acc[nt],
                                                        false, false);
    }
  }

  // ---- epilogue: normalize and store context (bf16 row-major [BT,NE]) ----
  float inv[8];
  #pragma unroll
  for (int j = 0; j < 8; ++j) inv[j] = 1.f / lrun[j];
  __bf16* crow = Ctx + (size_t)(b * TT + q0) * NE + h * EE;
  #pragma unroll
  for (int nt = 0; nt < 16; ++nt)
    #pragma unroll
    for (int j = 0; j < 8; ++j)
      crow[(size_t)(j + 8 * hi) * NE + nt * 16 + ln] =
          (__bf16)(acc[nt][j] * inv[j]);
}

// ---------------------------------------------------------------------------
// Kernel 3: output projection  out[BT,E] (f32) = Ctx[BT,NE] * Wo  (via Wot)
// grid: (BT/16=512, 2), block 256 (8 waves; wave owns one 16-col tile)
// ---------------------------------------------------------------------------
__global__ __launch_bounds__(256)
void proj_kernel(const __bf16* __restrict__ Ctx, const __bf16* __restrict__ Wot,
                 float* __restrict__ out) {
  const int lane = threadIdx.x & 31;
  const int wave = threadIdx.x >> 5;
  const int hi = lane >> 4, ln = lane & 15;
  const int m0 = blockIdx.x * 16;
  const int n0 = (blockIdx.y * 8 + wave) * 16;

  const __bf16* arow = Ctx + (size_t)(m0 + ln) * NE + (hi ? 16 : 0);
  const __bf16* brow = Wot + (size_t)(n0 + ln) * NE + (hi ? 16 : 0);

  v8f c = {};
  for (int k0 = 0; k0 < NE; k0 += 32) {
    v16bf a = load_afrag(arow, k0);
    v16bf b = *reinterpret_cast<const v16bf*>(brow + k0);
    c = __builtin_amdgcn_wmma_f32_16x16x32_bf16(false, a, false, b,
                                                (short)0, c, false, false);
  }
  #pragma unroll
  for (int j = 0; j < 8; ++j)
    out[(size_t)(m0 + j + 8 * hi) * EE + n0 + ln] = c[j];
}

// ---------------------------------------------------------------------------
extern "C" void kernel_launch(void* const* d_in, const int* in_sizes, int n_in,
                              void* d_out, int out_size, void* d_ws, size_t ws_size,
                              hipStream_t stream) {
  const float* x  = (const float*)d_in[0];
  const unsigned char* mask = (const unsigned char*)d_in[1];  // jax bool: 1 byte
  const float* Wq = (const float*)d_in[2];
  const float* Wk = (const float*)d_in[3];
  const float* Wv = (const float*)d_in[4];
  const float* Wo = (const float*)d_in[5];
  float* out = (float*)d_out;

  char* p = (char*)d_ws;
  __bf16* xb  = (__bf16*)p;  p += (size_t)BT * EE * 2;        //  4 MB
  __bf16* Wt  = (__bf16*)p;  p += (size_t)3 * NE * EE * 2;    //  3 MB
  __bf16* Wot = (__bf16*)p;  p += (size_t)EE * NE * 2;        //  1 MB
  __bf16* Qb  = (__bf16*)p;  p += (size_t)BT * NE * 2;        // 32 MB
  __bf16* Kb  = (__bf16*)p;  p += (size_t)BT * NE * 2;        // 32 MB
  __bf16* Vt  = (__bf16*)p;  p += (size_t)BT * NE * 2;        // 32 MB
  __bf16* Ctx = (__bf16*)p;                                   // 32 MB

  hipLaunchKernelGGL(prep_kernel, dim3(8192, 5), dim3(256), 0, stream,
                     x, Wq, Wk, Wv, Wo, xb, Wt, Wot);
  hipLaunchKernelGGL(qkv_kernel, dim3(512, 16, 3), dim3(256), 0, stream,
                     xb, Wt, Qb, Kb, Vt);
  hipLaunchKernelGGL(attn_kernel, dim3(64, 8, 8), dim3(32), 0, stream,
                     Qb, Kb, Vt, mask, Ctx);
  hipLaunchKernelGGL(proj_kernel, dim3(512, 2), dim3(256), 0, stream,
                     Ctx, Wot, out);
}
